// RoPEAttentionHead_55301998903547
// MI455X (gfx1250) — matrix operand order repaired and checked by reading, not compiled, not run on hardware
//
#include <hip/hip_runtime.h>

#define D      128
#define SEQL   4096
#define BATCHN 4

typedef __attribute__((ext_vector_type(16))) _Float16 v16h;
typedef __attribute__((ext_vector_type(8)))  _Float16 v8h;
typedef __attribute__((ext_vector_type(8)))  float    v8f;

static __device__ inline v8f wmma_f16(v16h a, v16h b, v8f c) {
  // D = A(16x32 f16) * B(32x16 f16) + C(16x16 f32)
  return __builtin_amdgcn_wmma_f32_16x16x32_f16(false, a, false, b, (short)0, c,
                                                false, false);
}

// A-layout fragment (16x32, f16) from row-major matrix, ld in halves.
// Lane l: row = l&15; halves 0-7 at K = (l>>4)*8, halves 8-15 at K = 16+(l>>4)*8.
static __device__ inline v16h load_fragA_h(const _Float16* tile, int ld, int lane) {
  int row = lane & 15, hi = lane >> 4;
  const _Float16* p = tile + row * ld + hi * 8;
  v16h a;
  ((v8h*)&a)[0] = *(const v8h*)(p);
  ((v8h*)&a)[1] = *(const v8h*)(p + 16);
  return a;
}

// B-layout fragment (32x16, f16): B[k][n] = mem[n*ld + k].
// Lane l: col n = l&15; halves 0-7 at K = (l>>4)*8, halves 8-15 at K = 16+(l>>4)*8.
static __device__ inline v16h load_fragB_h(const _Float16* tile, int ld, int lane) {
  int col = lane & 15, hi = lane >> 4;
  const _Float16* p = tile + col * ld + hi * 8;
  v16h b;
  ((v8h*)&b)[0] = *(const v8h*)(p);
  ((v8h*)&b)[1] = *(const v8h*)(p + 16);
  return b;
}

// ---------------------------------------------------------------------------
// Kernel 0: extract the 2 non-trivial entries per rotation pair from the dense
// 256MB R tensor into a compact 2MB (cos,sin) table.
// ---------------------------------------------------------------------------
__global__ void rope_cs_kernel(const float* __restrict__ R, float2* __restrict__ cs) {
  int t = blockIdx.x * blockDim.x + threadIdx.x;
  if (t >= SEQL * (D / 2)) return;
  int m = t >> 6;   // position
  int i = t & 63;   // pair index
  const float* Rm = R + (size_t)m * D * D;
  float c = Rm[(2 * i) * D + 2 * i];       // cos
  float s = Rm[(2 * i + 1) * D + 2 * i];   // sin
  cs[t] = make_float2(c, s);
}

// ---------------------------------------------------------------------------
// Kernel 1: q/k/v projections via WMMA + RoPE, f16 outputs.
// One wave per 16-row tile. Weights staged as f16 in dynamic LDS (96KB).
// qr,kr: [B,SEQ,128] row-major.  vt: [B,128,SEQ] (V transposed for PV B-frags).
// ---------------------------------------------------------------------------
__global__ __launch_bounds__(128) void proj_rope_kernel(
    const float* __restrict__ x, const float* __restrict__ Wq,
    const float* __restrict__ Wk, const float* __restrict__ Wv,
    const float2* __restrict__ cs,
    _Float16* __restrict__ qr, _Float16* __restrict__ kr,
    _Float16* __restrict__ vt) {
  extern __shared__ _Float16 sW[];  // 3 * 128*128 halves
  int tid = threadIdx.x;
  for (int idx = tid; idx < D * D; idx += 128) sW[idx]             = (_Float16)Wq[idx];
  for (int idx = tid; idx < D * D; idx += 128) sW[D * D + idx]     = (_Float16)Wk[idx];
  for (int idx = tid; idx < D * D; idx += 128) sW[2 * D * D + idx] = (_Float16)Wv[idx];
  __syncthreads();

  int wave = tid >> 5, lane = tid & 31;
  int gw = blockIdx.x * 4 + wave;          // 0..1023
  int b = gw >> 8, tile = gw & 255;        // 256 row-tiles per batch
  int m0 = tile * 16;
  int row = lane & 15, hi = lane >> 4;

  // A fragments of x (f32 -> f16 on the fly)
  const float* xrow = x + ((size_t)(b * SEQL + m0 + row)) * D;
  v16h ax[4];
#pragma unroll
  for (int c = 0; c < 4; ++c) {
    const float* p = xrow + c * 32 + hi * 8;
    v16h a;
#pragma unroll
    for (int j = 0; j < 8; ++j) {
      a[j]     = (_Float16)p[j];
      a[8 + j] = (_Float16)p[16 + j];
    }
    ax[c] = a;
  }

#pragma unroll
  for (int et = 0; et < 8; ++et) {
    v8f aq = {}, ak = {}, av = {};
#pragma unroll
    for (int c = 0; c < 4; ++c) {
      // q[m,e] = sum_d x[m,d]*W[e,d]  ->  B[d,e] = W[e,d], k-contig per lane
      v16h bq = load_fragB_h(sW             + (et * 16) * D + c * 32, D, lane);
      v16h bk = load_fragB_h(sW + D * D     + (et * 16) * D + c * 32, D, lane);
      v16h bv = load_fragB_h(sW + 2 * D * D + (et * 16) * D + c * 32, D, lane);
      aq = wmma_f16(ax[c], bq, aq);
      ak = wmma_f16(ax[c], bk, ak);
      av = wmma_f16(ax[c], bv, av);
    }
    // RoPE: e even: c*q[e] + s*q[e+1] ; e odd: c*q[e] - s*q[e-1]
    int e = et * 16 + row;
    int i = e >> 1;
    bool even = (e & 1) == 0;
#pragma unroll
    for (int r = 0; r < 8; ++r) {
      int mrow = m0 + r + hi * 8;
      float2 csv = cs[mrow * 64 + i];
      float qv = aq[r], kv = ak[r];
      float qp = __shfl_xor(qv, 1, 32);  // partner element e^1 lives in lane^1
      float kp = __shfl_xor(kv, 1, 32);
      float qo = even ? (csv.x * qv + csv.y * qp) : (csv.x * qv - csv.y * qp);
      float ko = even ? (csv.x * kv + csv.y * kp) : (csv.x * kv - csv.y * kp);
      size_t ro = ((size_t)(b * SEQL + mrow)) * D + e;
      qr[ro] = (_Float16)qo;
      kr[ro] = (_Float16)ko;
      vt[((size_t)(b * D + e)) * SEQL + mrow] = (_Float16)av[r];
    }
  }
}

// ---------------------------------------------------------------------------
// Kernel 2: causal flash attention, key-split 4 ways for occupancy.
// One block (4 waves, 128 threads) per 16-query tile; wave w handles key
// blocks n0 = w*32, w*32+128, ... (interleaved for causal load balance).
// Per 32-key block: 8 WMMAs (S), base-2 online softmax with shfl_xor row
// reductions, P transposed C-layout -> A-layout via per-wave LDS tile,
// 8 WMMAs (PV). Partials merged across waves via LDS log-sum-exp combine.
// ---------------------------------------------------------------------------
__global__ __launch_bounds__(128) void attn_kernel(
    const _Float16* __restrict__ qr, const _Float16* __restrict__ kr,
    const _Float16* __restrict__ vt, float* __restrict__ out) {
  __shared__ __align__(16) _Float16 ptile[4][16 * 32];
  __shared__ __align__(16) float so[3][16][D];  // O partials of waves 1..3
  __shared__ float sm[3][16];                   // row max of waves 1..3
  __shared__ float sl[3][16];                   // row sum of waves 1..3
  int tid = threadIdx.x;
  int wave = tid >> 5, lane = tid & 31;
  int b = blockIdx.x >> 8, tile = blockIdx.x & 255;
  int q0 = tile * 16;
  int row = lane & 15, hi = lane >> 4;
  const _Float16* qb = qr + (size_t)b * SEQL * D;
  const _Float16* kb = kr + (size_t)b * SEQL * D;
  const _Float16* vb = vt + (size_t)b * D * SEQL;

  v16h aq[4];
#pragma unroll
  for (int c = 0; c < 4; ++c)
    aq[c] = load_fragA_h(qb + (size_t)q0 * D + c * 32, D, lane);

  v8f zero8 = {};
  v8f o[8];
#pragma unroll
  for (int et = 0; et < 8; ++et) o[et] = zero8;
  float m2[8], l[8];
#pragma unroll
  for (int r = 0; r < 8; ++r) { m2[r] = -__builtin_inff(); l[r] = 0.f; }

  const float csc = 0.08838834764831845f * 1.4426950408889634f;  // 1/sqrt(128)*log2(e)
  _Float16* pt = ptile[wave];
  int nk = q0 + 16;  // causal: keys 0..q0+15

  for (int n0 = wave * 32; n0 < nk; n0 += 128) {
    v8f s0 = zero8, s1 = zero8;
#pragma unroll
    for (int c = 0; c < 4; ++c) {
      // S[m,n] = sum_d q[m,d]*k[n,d] -> B[d,n] = k_rot[n,d], k-contig per lane
      v16h bk0 = load_fragB_h(kb + (size_t)n0 * D + c * 32, D, lane);
      v16h bk1 = load_fragB_h(kb + (size_t)(n0 + 16) * D + c * 32, D, lane);
      s0 = wmma_f16(aq[c], bk0, s0);
      s1 = wmma_f16(aq[c], bk1, s1);
    }
    bool need_mask = (n0 + 31 > q0);
    float p0[8], p1[8], rmax[8];
#pragma unroll
    for (int r = 0; r < 8; ++r) {
      float a0 = s0[r] * csc, a1 = s1[r] * csc;
      if (need_mask) {
        int qm = q0 + r + hi * 8;
        if (n0 + row > qm)      a0 = -__builtin_inff();
        if (n0 + 16 + row > qm) a1 = -__builtin_inff();
      }
      p0[r] = a0; p1[r] = a1;
      rmax[r] = fmaxf(a0, a1);
    }
    // row max across the 16 lanes of each half-wave (C-layout: lane = column)
#pragma unroll
    for (int off = 1; off < 16; off <<= 1) {
#pragma unroll
      for (int r = 0; r < 8; ++r)
        rmax[r] = fmaxf(rmax[r], __shfl_xor(rmax[r], off, 32));
    }
    float alpha[8], rs[8];
#pragma unroll
    for (int r = 0; r < 8; ++r) {
      float mn = fmaxf(m2[r], rmax[r]);
      alpha[r] = exp2f(m2[r] - mn);  // first block: exp2(-inf - finite) = 0
      m2[r] = mn;
      p0[r] = exp2f(p0[r] - mn);     // masked: exp2(-inf) = 0
      p1[r] = exp2f(p1[r] - mn);
      rs[r] = p0[r] + p1[r];
    }
#pragma unroll
    for (int off = 1; off < 16; off <<= 1) {
#pragma unroll
      for (int r = 0; r < 8; ++r)
        rs[r] += __shfl_xor(rs[r], off, 32);
    }
#pragma unroll
    for (int r = 0; r < 8; ++r) l[r] = l[r] * alpha[r] + rs[r];

    // transpose P (C-layout) -> A-layout via per-wave LDS tile (16x32 f16)
#pragma unroll
    for (int r = 0; r < 8; ++r) {
      pt[(r + hi * 8) * 32 + row]      = (_Float16)p0[r];
      pt[(r + hi * 8) * 32 + 16 + row] = (_Float16)p1[r];
    }
    asm volatile("s_wait_dscnt 0" ::: "memory");  // same-wave DS RAW (defensive)
    v16h ap = load_fragA_h(pt, 32, lane);

#pragma unroll
    for (int et = 0; et < 8; ++et) {
      v8f t = o[et];
#pragma unroll
      for (int r = 0; r < 8; ++r) t[r] *= alpha[r];
      // O[m,e] += sum_n P[m,n]*V[n,e] -> B[n,e] = vt[e, n0+n], k-contig per lane
      v16h bv = load_fragB_h(vb + (size_t)(et * 16) * SEQL + n0, SEQL, lane);
      o[et] = wmma_f16(ap, bv, t);
    }
  }

  // ---- cross-wave log-sum-exp merge of the 4 key-split partials ----
  if (wave != 0) {
    if (row == 0) {  // lanes 0 (rows 0-7) and 16 (rows 8-15) hold the stats
#pragma unroll
      for (int r = 0; r < 8; ++r) {
        sm[wave - 1][hi * 8 + r] = m2[r];
        sl[wave - 1][hi * 8 + r] = l[r];
      }
    }
#pragma unroll
    for (int et = 0; et < 8; ++et)
#pragma unroll
      for (int r = 0; r < 8; ++r)
        so[wave - 1][r + hi * 8][et * 16 + row] = o[et][r];
  }
  __syncthreads();
  if (wave == 0) {
    float M[8], L[8], beta[8];
#pragma unroll
    for (int r = 0; r < 8; ++r) {
      M[r] = m2[r];  // wave 0 always saw key block 0 -> finite
#pragma unroll
      for (int w = 0; w < 3; ++w) M[r] = fmaxf(M[r], sm[w][r + hi * 8]);
      beta[r] = exp2f(m2[r] - M[r]);
      L[r] = l[r] * beta[r];
    }
#pragma unroll
    for (int et = 0; et < 8; ++et)
#pragma unroll
      for (int r = 0; r < 8; ++r) o[et][r] *= beta[r];
#pragma unroll
    for (int w = 0; w < 3; ++w) {
      float bw[8];
#pragma unroll
      for (int r = 0; r < 8; ++r) {
        bw[r] = exp2f(sm[w][r + hi * 8] - M[r]);  // idle wave: exp2(-inf) = 0
        L[r] += sl[w][r + hi * 8] * bw[r];
      }
#pragma unroll
      for (int et = 0; et < 8; ++et)
#pragma unroll
        for (int r = 0; r < 8; ++r)
          o[et][r] += so[w][r + hi * 8][et * 16 + row] * bw[r];
    }
    float inv[8];
#pragma unroll
    for (int r = 0; r < 8; ++r) inv[r] = 1.0f / L[r];
    float* ob = out + ((size_t)b * SEQL + q0) * D;
#pragma unroll
    for (int et = 0; et < 8; ++et) {
#pragma unroll
      for (int r = 0; r < 8; ++r)
        ob[(size_t)(r + hi * 8) * D + et * 16 + row] = o[et][r] * inv[r];
    }
  }
}

// ---------------------------------------------------------------------------
// Workspace layout (bytes):
//   [0,   4MiB)  qr  f16 [B,SEQ,128]
//   [4,   8MiB)  kr  f16 [B,SEQ,128]
//   [8,  12MiB)  vt  f16 [B,128,SEQ]
//   [12, 14MiB)  cs  float2 [SEQ,64]
// ---------------------------------------------------------------------------
extern "C" void kernel_launch(void* const* d_in, const int* in_sizes, int n_in,
                              void* d_out, int out_size, void* d_ws, size_t ws_size,
                              hipStream_t stream) {
  const float* x  = (const float*)d_in[0];
  const float* Wq = (const float*)d_in[1];
  const float* Wk = (const float*)d_in[2];
  const float* Wv = (const float*)d_in[3];
  const float* R  = (const float*)d_in[4];
  float* out = (float*)d_out;

  char* ws = (char*)d_ws;
  _Float16* qr = (_Float16*)(ws);
  _Float16* kr = (_Float16*)(ws + ((size_t)1 << 22));
  _Float16* vt = (_Float16*)(ws + ((size_t)2 << 22));
  float2*   cs = (float2*)(ws + ((size_t)3 << 22));

  rope_cs_kernel<<<(SEQL * 64 + 255) / 256, 256, 0, stream>>>(R, cs);
  proj_rope_kernel<<<256, 128, 3 * D * D * sizeof(_Float16), stream>>>(
      x, Wq, Wk, Wv, cs, qr, kr, vt);
  attn_kernel<<<BATCHN * 256, 128, 0, stream>>>(qr, kr, vt, out);
}
